// LSTM_76922864271739
// MI455X (gfx1250) — compile-verified, matrix operand order checked by reading, not run on hardware
//
#include <hip/hip_runtime.h>
#include <hip/hip_bf16.h>
#include <math.h>

#define T_SEQ 2048
#define I_DIM 1024
#define H_DIM 2048
#define O_DIM 512
#define HI_DIM 3072

typedef __bf16 bf16;
typedef __attribute__((ext_vector_type(16))) __bf16 v16bf;
typedef __attribute__((ext_vector_type(8)))  __bf16 v8bf;
typedef __attribute__((ext_vector_type(8)))  float  v8f;

// Load 16 contiguous bf16 as two 16-byte vector loads (global_load_b128 pairs).
__device__ __forceinline__ v16bf load_frag16(const bf16* p) {
    union { v16bf v; v8bf h[2]; } u;
    u.h[0] = *reinterpret_cast<const v8bf*>(p);
    u.h[1] = *reinterpret_cast<const v8bf*>(p + 8);
    return u.v;
}

__device__ __forceinline__ float sigmoidf_(float x) {
    return 1.0f / (1.0f + __expf(-x));
}

// ---------------------------------------------------------------------------
// Init: zero h_{-1} row (row 0 of hball) and the grid-barrier counter.
// Runs every launch so graph replays are deterministic.
// ---------------------------------------------------------------------------
__global__ void init_kernel(bf16* __restrict__ hball, unsigned* __restrict__ cnt) {
    int i = blockIdx.x * blockDim.x + threadIdx.x;
    if (i < H_DIM) hball[i] = (bf16)0.0f;
    if (i == 0) *cnt = 0u;
}

// ---------------------------------------------------------------------------
// Convert the 4 gate weight matrices (H x (H+I), fp32) into split bf16 copies:
// Whb[g][m][0..H) (recurrent part) and Wxb[g][m][0..I) (input part).
// ---------------------------------------------------------------------------
__global__ void conv_w_kernel(const float* __restrict__ Wf, const float* __restrict__ Wi,
                              const float* __restrict__ Wc, const float* __restrict__ Wo,
                              bf16* __restrict__ Whb, bf16* __restrict__ Wxb) {
    const long per_gate = (long)H_DIM * HI_DIM;
    const long total = 4L * per_gate;
    for (long idx = blockIdx.x * (long)blockDim.x + threadIdx.x; idx < total;
         idx += (long)gridDim.x * blockDim.x) {
        int g = (int)(idx / per_gate);
        long rem = idx % per_gate;
        int row = (int)(rem / HI_DIM);
        int col = (int)(rem % HI_DIM);
        const float* W = (g == 0) ? Wf : (g == 1) ? Wi : (g == 2) ? Wc : Wo;
        float v = W[(long)row * HI_DIM + col];
        if (col < H_DIM)
            Whb[((long)g * H_DIM + row) * H_DIM + col] = (bf16)v;
        else
            Wxb[((long)g * H_DIM + row) * I_DIM + (col - H_DIM)] = (bf16)v;
    }
}

__global__ void conv_plain_kernel(const float* __restrict__ src, bf16* __restrict__ dst, long n) {
    for (long i = blockIdx.x * (long)blockDim.x + threadIdx.x; i < n;
         i += (long)gridDim.x * blockDim.x)
        dst[i] = (bf16)src[i];
}

// ---------------------------------------------------------------------------
// Batched input-contribution GEMM (no recurrence):
//   P[g][t][m] = sum_k Wxb[g][m][k] * Xb[t][k]
// One wave computes one 16(m) x 16(t) tile of one gate; K = 1024 in 32 chunks.
// ---------------------------------------------------------------------------
__global__ void __launch_bounds__(256) gemm_input_kernel(const bf16* __restrict__ Wxb,
                                                         const bf16* __restrict__ Xb,
                                                         float* __restrict__ P) {
    const int wave = threadIdx.x >> 5;
    const int lane = threadIdx.x & 31;
    const int w = blockIdx.x * 8 + wave;          // 0 .. 65535
    const int g  = w >> 14;                        // / 16384
    const int rem = w & 16383;
    const int mt = rem >> 7;                       // 0..127
    const int tt = rem & 127;                      // 0..127
    const int m0 = mt * 16, t0 = tt * 16;
    const int row = lane & 15;
    const int kh  = (lane >> 4) * 16;

    const bf16* aBase = Wxb + ((long)g * H_DIM + m0 + row) * I_DIM + kh;
    const bf16* bBase = Xb + (long)(t0 + row) * I_DIM + kh;

    v8f acc = {};
    for (int kc = 0; kc < I_DIM / 32; ++kc) {
        v16bf a = load_frag16(aBase + kc * 32);
        v16bf b = load_frag16(bBase + kc * 32);
        acc = __builtin_amdgcn_wmma_f32_16x16x32_bf16(false, a, false, b,
                                                      (short)0, acc, false, false);
    }

    const int n  = lane & 15;
    const int mb = (lane >> 4) * 8;
    float* dst = P + ((long)g * T_SEQ + t0 + n) * H_DIM + m0 + mb;
#pragma unroll
    for (int r = 0; r < 8; ++r) dst[r] = acc[r];
}

// ---------------------------------------------------------------------------
// Persistent recurrence kernel. 128 blocks x 256 threads; block b owns rows
// [16b, 16b+16). 8 waves = 4 gates x 2 K-halves. Per step each wave does a
// 16x16x2048-half GEMV tile via WMMA with h replicated across B columns,
// then lanes 0/16 publish the (identical) column to LDS; threads 0..15 fuse
// gates + cell update and write h_t (fp32 to output, bf16 for next step).
// Grid-wide sync: monotonically increasing atomic counter + s_sleep spin.
// ---------------------------------------------------------------------------
__global__ void __launch_bounds__(256) lstm_rec_kernel(
    const bf16* __restrict__ Whb, const float* __restrict__ P,
    const float* __restrict__ bf_, const float* __restrict__ bi_,
    const float* __restrict__ bc_, const float* __restrict__ bo_,
    float* __restrict__ hid_out, bf16* __restrict__ hball,
    unsigned* __restrict__ cnt) {
    __shared__ float gacc[8][16];

    const int wave = threadIdx.x >> 5;
    const int lane = threadIdx.x & 31;
    const int gate = wave & 3;
    const int ksel = wave >> 2;                    // 0 -> K[0,1024), 1 -> K[1024,2048)
    const int m0 = blockIdx.x * 16;
    const int row = lane & 15;
    const int kh  = (lane >> 4) * 16;
    const unsigned nblocks = gridDim.x;

    const bf16* aBase = Whb + ((long)gate * H_DIM + m0 + row) * H_DIM + ksel * 1024 + kh;

    float creg = 0.0f;                             // cell state for row m0+tid (tid<16)

    for (int t = 0; t < T_SEQ; ++t) {
        const bf16* hrow = hball + (long)t * H_DIM + ksel * 1024 + kh;

        v8f acc = {};
#pragma unroll 4
        for (int kc = 0; kc < 32; ++kc) {
            v16bf a = load_frag16(aBase + kc * 32);
            v16bf b = load_frag16(hrow + kc * 32);   // h replicated across B columns
            acc = __builtin_amdgcn_wmma_f32_16x16x32_bf16(false, a, false, b,
                                                          (short)0, acc, false, false);
        }

        // All B columns are identical: lane 0 holds rows m0..m0+7 in acc[0..7],
        // lane 16 holds rows m0+8..m0+15.
        if (lane == 0) {
#pragma unroll
            for (int r = 0; r < 8; ++r) gacc[wave][r] = acc[r];
        } else if (lane == 16) {
#pragma unroll
            for (int r = 0; r < 8; ++r) gacc[wave][8 + r] = acc[r];
        }
        __syncthreads();

        if (threadIdx.x < 16) {
            const int tid = threadIdx.x;
            const int m = m0 + tid;
            const long pb = (long)t * H_DIM + m;
            float fp = gacc[0][tid] + gacc[4][tid] + P[0L * T_SEQ * H_DIM + pb] + bf_[m];
            float ip = gacc[1][tid] + gacc[5][tid] + P[1L * T_SEQ * H_DIM + pb] + bi_[m];
            float cp = gacc[2][tid] + gacc[6][tid] + P[2L * T_SEQ * H_DIM + pb] + bc_[m];
            float op = gacc[3][tid] + gacc[7][tid] + P[3L * T_SEQ * H_DIM + pb] + bo_[m];
            float f  = sigmoidf_(fp);
            float ii = sigmoidf_(ip);
            float cc = tanhf(cp);
            float oo = sigmoidf_(op);
            creg = f * creg + ii * cc;
            float h = oo * tanhf(creg);
            hid_out[(long)t * H_DIM + m] = h;
            hball[(long)(t + 1) * H_DIM + m] = (bf16)h;
        }
        __threadfence();
        __syncthreads();

        // Grid-wide barrier: counter only ever increases within one launch.
        if (threadIdx.x == 0) {
            __hip_atomic_fetch_add(cnt, 1u, __ATOMIC_ACQ_REL, __HIP_MEMORY_SCOPE_AGENT);
            const unsigned target = (unsigned)(t + 1) * nblocks;
            while (__hip_atomic_load(cnt, __ATOMIC_ACQUIRE, __HIP_MEMORY_SCOPE_AGENT) < target)
                __builtin_amdgcn_s_sleep(1);
        }
        __syncthreads();
    }
}

// ---------------------------------------------------------------------------
// Batched output projection (no recurrence): Y[t][o] = W_y[o][:] . h_t + b_y[o]
// B columns are rows 1..T of hball (bf16 hidden history).
// ---------------------------------------------------------------------------
__global__ void __launch_bounds__(256) gemm_y_kernel(const bf16* __restrict__ Wyb,
                                                     const bf16* __restrict__ hball,
                                                     const float* __restrict__ by,
                                                     float* __restrict__ Yout) {
    const int wave = threadIdx.x >> 5;
    const int lane = threadIdx.x & 31;
    const int w = blockIdx.x * 8 + wave;           // 0 .. 4095
    const int ot = w >> 7;                         // 0..31
    const int tt = w & 127;                        // 0..127
    const int o0 = ot * 16, t0 = tt * 16;
    const int row = lane & 15;
    const int kh  = (lane >> 4) * 16;

    const bf16* aBase = Wyb + (long)(o0 + row) * H_DIM + kh;
    const bf16* bBase = hball + (long)(t0 + row + 1) * H_DIM + kh;

    v8f acc = {};
    for (int kc = 0; kc < H_DIM / 32; ++kc) {
        v16bf a = load_frag16(aBase + kc * 32);
        v16bf b = load_frag16(bBase + kc * 32);
        acc = __builtin_amdgcn_wmma_f32_16x16x32_bf16(false, a, false, b,
                                                      (short)0, acc, false, false);
    }

    const int n  = lane & 15;
    const int mb = (lane >> 4) * 8;
    float* dst = Yout + (long)(t0 + n) * O_DIM + o0 + mb;
#pragma unroll
    for (int r = 0; r < 8; ++r) dst[r] = acc[r] + by[o0 + mb + r];
}

// ---------------------------------------------------------------------------
extern "C" void kernel_launch(void* const* d_in, const int* in_sizes, int n_in,
                              void* d_out, int out_size, void* d_ws, size_t ws_size,
                              hipStream_t stream) {
    const float* X   = (const float*)d_in[0];
    const float* Wf  = (const float*)d_in[1];
    const float* bf_ = (const float*)d_in[2];
    const float* Wi  = (const float*)d_in[3];
    const float* bi  = (const float*)d_in[4];
    const float* Wc  = (const float*)d_in[5];
    const float* bc  = (const float*)d_in[6];
    const float* Wo  = (const float*)d_in[7];
    const float* bo  = (const float*)d_in[8];
    const float* Wy  = (const float*)d_in[9];
    const float* by  = (const float*)d_in[10];

    float* Yout = (float*)d_out;                               // (T, O)
    float* Hout = (float*)d_out + (long)T_SEQ * O_DIM;         // (T, H)

    // Workspace carve-up (256B aligned), ~126 MB total.
    char* ws = (char*)d_ws;
    size_t off = 0;
    auto carve = [&](size_t bytes) -> void* {
        void* p = ws + off;
        off = (off + bytes + 255) & ~(size_t)255;
        return p;
    };
    bf16*     Whb   = (bf16*)carve(4L * H_DIM * H_DIM * sizeof(bf16));   // 33.5 MB
    bf16*     Wxb   = (bf16*)carve(4L * H_DIM * I_DIM * sizeof(bf16));   // 16.8 MB
    bf16*     Wyb   = (bf16*)carve((long)O_DIM * H_DIM * sizeof(bf16));  //  2.1 MB
    bf16*     Xb    = (bf16*)carve((long)T_SEQ * I_DIM * sizeof(bf16));  //  4.2 MB
    float*    P     = (float*)carve(4L * T_SEQ * H_DIM * sizeof(float)); // 67.1 MB
    bf16*     hball = (bf16*)carve((long)(T_SEQ + 1) * H_DIM * sizeof(bf16)); // 8.4 MB
    unsigned* cnt   = (unsigned*)carve(256);

    // 1) deterministic per-launch init (h_{-1}=0, barrier counter=0)
    init_kernel<<<(H_DIM + 255) / 256, 256, 0, stream>>>(hball, cnt);

    // 2) fp32 -> bf16 conversions
    conv_w_kernel<<<4096, 256, 0, stream>>>(Wf, Wi, Wc, Wo, Whb, Wxb);
    conv_plain_kernel<<<2048, 256, 0, stream>>>(X, Xb, (long)T_SEQ * I_DIM);
    conv_plain_kernel<<<1024, 256, 0, stream>>>(Wy, Wyb, (long)O_DIM * H_DIM);

    // 3) batched input-contribution GEMM: 4*128*128 wave tiles / 8 waves per block
    gemm_input_kernel<<<8192, 256, 0, stream>>>(Wxb, Xb, P);

    // 4) sequential recurrence (persistent, grid-synced)
    lstm_rec_kernel<<<128, 256, 0, stream>>>(Whb, P, bf_, bi, bc, bo, Hout, hball, cnt);

    // 5) batched output projection: 32*128 wave tiles / 8 waves per block
    gemm_y_kernel<<<512, 256, 0, stream>>>(Wyb, hball, by, Yout);
}
